// SequenceLabelingMamba_73212012527873
// MI455X (gfx1250) — compile-verified, matrix operand order checked by reading, not run on hardware
//
#include <hip/hip_runtime.h>
#include <hip/hip_bf16.h>
#include <cstdint>
#include <cstddef>

typedef __bf16 bf16_t;
typedef __attribute__((ext_vector_type(16))) __bf16 v16bf;
typedef __attribute__((ext_vector_type(8)))  __bf16 v8bf;
typedef __attribute__((ext_vector_type(8)))  float   v8f;

#define B_SZ     8
#define T_LEN    4096
#define BT       (B_SZ * T_LEN)     // 32768 token rows
#define D_MODEL  256
#define D_INNER  512
#define D_STATE  16
#define DT_RANK  16
#define N_LAYERS 4
#define N_CLASSES 32
#define XD       (DT_RANK + 2 * D_STATE)   // 48

// Async global->LDS 16B copy (CDNA5, ASYNCcnt-tracked). GVS addressing:
// lds byte-offset VGPR, 32-bit per-lane byte offset VGPR, 64-bit SGPR base.
__device__ __forceinline__ void async_ld_b128(unsigned lds_addr, unsigned voff,
                                              unsigned long long sbase)
{
    asm volatile("global_load_async_to_lds_b128 %0, %1, %2"
                 :: "v"(lds_addr), "v"(voff), "s"(sbase) : "memory");
}
__device__ __forceinline__ void wait_asynccnt0()
{
    asm volatile("s_wait_asynccnt 0x0" ::: "memory");
}

// ---------------------------------------------------------------------------
// fp32 -> bf16 conversion (weights + activations feeding WMMA)
// ---------------------------------------------------------------------------
__global__ __launch_bounds__(256) void cvt_f32_bf16(const float* __restrict__ src,
                                                    bf16_t* __restrict__ dst, int n)
{
    int i = blockIdx.x * 256 + threadIdx.x;
    if (i < n) dst[i] = (bf16_t)src[i];
}

// ---------------------------------------------------------------------------
// GEMM: C[M,N] (f32) = A[M,K] (bf16, row major) x W[N,K]^T (bf16, row major)
//       optional bias[N], optional C += (residual accumulate)
// Block tile 128x64, K-step 32, 8 waves (wave32), each wave -> 2x2 WMMA tiles.
// NGUARD=false: async global->LDS double-buffered pipeline (N % 64 == 0).
// NGUARD=true : synchronous path with N bounds checks + zero fill.
// ---------------------------------------------------------------------------
template <bool NGUARD>
__global__ __launch_bounds__(256)
void gemm_bf16_wmma(const bf16_t* __restrict__ A, const bf16_t* __restrict__ W,
                    float* __restrict__ C, const float* __restrict__ bias,
                    int M, int N, int K, int accumulate)
{
    constexpr int BM = 128, BN = 64, BK = 32, LSTR = 40; // 80B row stride: conflict-free
    __shared__ bf16_t As[2][BM * LSTR];
    __shared__ bf16_t Ws[2][BN * LSTR];

    const int m0   = blockIdx.y * BM;
    const int n0   = blockIdx.x * BN;
    const int tid  = threadIdx.x;
    const int lane = tid & 31;
    const int wave = tid >> 5;
    const int wm   = wave >> 1;           // 0..3 : 32-row strip
    const int wn   = wave & 1;            // 0..1 : 32-col strip
    const int fm   = lane & 15;           // fragment row/col
    const int kb   = (lane >> 4) << 3;    // 0 or 8 (K sub-block per lane half)

    v8f acc00 = {}, acc01 = {}, acc10 = {}, acc11 = {};

    union Frag { v16bf v; v8bf h[2]; };

    const int row = tid >> 2;             // 0..63
    const int kc  = (tid & 3) << 3;       // 0,8,16,24

    if constexpr (!NGUARD) {
        // --- async double-buffered pipeline -------------------------------
        const unsigned long long Ab = (unsigned long long)(uintptr_t)A;
        const unsigned long long Wb = (unsigned long long)(uintptr_t)W;
        const unsigned aoff0 = (unsigned)(((m0 + row) * K + kc) * 2);
        const unsigned aoff1 = (unsigned)(((m0 + row + 64) * K + kc) * 2);
        const unsigned woff  = (unsigned)(((n0 + row) * K + kc) * 2);
        unsigned lA0[2], lA1[2], lWr[2];
#pragma unroll
        for (int bfr = 0; bfr < 2; ++bfr) {
            lA0[bfr] = (unsigned)(uintptr_t)&As[bfr][row * LSTR + kc];
            lA1[bfr] = (unsigned)(uintptr_t)&As[bfr][(row + 64) * LSTR + kc];
            lWr[bfr] = (unsigned)(uintptr_t)&Ws[bfr][row * LSTR + kc];
        }
        // issue tile 0 into buffer 0
        async_ld_b128(lA0[0], aoff0, Ab);
        async_ld_b128(lA1[0], aoff1, Ab);
        async_ld_b128(lWr[0], woff, Wb);

        const int ktiles = K / BK;
        int cur = 0;
        for (int kt = 0; kt < ktiles; ++kt) {
            wait_asynccnt0();        // own async copies of tile kt landed in LDS
            __syncthreads();         // everyone's landed; prev reads of cur^1 done
            if (kt + 1 < ktiles) {   // prefetch tile kt+1 into the other buffer
                unsigned d = (unsigned)((kt + 1) * BK * 2);
                async_ld_b128(lA0[cur ^ 1], aoff0 + d, Ab);
                async_ld_b128(lA1[cur ^ 1], aoff1 + d, Ab);
                async_ld_b128(lWr[cur ^ 1], woff + d, Wb);
            }
            Frag a0, a1, b0, b1;
            {
                int r = wm * 32 + fm;
                a0.h[0] = *(const v8bf*)(&As[cur][r * LSTR + kb]);
                a0.h[1] = *(const v8bf*)(&As[cur][r * LSTR + 16 + kb]);
                r += 16;
                a1.h[0] = *(const v8bf*)(&As[cur][r * LSTR + kb]);
                a1.h[1] = *(const v8bf*)(&As[cur][r * LSTR + 16 + kb]);
            }
            {
                int r = wn * 32 + fm;
                b0.h[0] = *(const v8bf*)(&Ws[cur][r * LSTR + kb]);
                b0.h[1] = *(const v8bf*)(&Ws[cur][r * LSTR + 16 + kb]);
                r += 16;
                b1.h[0] = *(const v8bf*)(&Ws[cur][r * LSTR + kb]);
                b1.h[1] = *(const v8bf*)(&Ws[cur][r * LSTR + 16 + kb]);
            }
            acc00 = __builtin_amdgcn_wmma_f32_16x16x32_bf16(false, a0.v, false, b0.v,
                                                            (short)0, acc00, false, false);
            acc01 = __builtin_amdgcn_wmma_f32_16x16x32_bf16(false, a0.v, false, b1.v,
                                                            (short)0, acc01, false, false);
            acc10 = __builtin_amdgcn_wmma_f32_16x16x32_bf16(false, a1.v, false, b0.v,
                                                            (short)0, acc10, false, false);
            acc11 = __builtin_amdgcn_wmma_f32_16x16x32_bf16(false, a1.v, false, b1.v,
                                                            (short)0, acc11, false, false);
            cur ^= 1;
        }
    } else {
        // --- synchronous path with N bounds guard -------------------------
        const bool wok = (n0 + row) < N;
        for (int k0 = 0; k0 < K; k0 += BK) {
#pragma unroll
            for (int i = 0; i < 2; ++i) {
                *(v8bf*)(&As[0][(row + 64 * i) * LSTR + kc]) =
                    *(const v8bf*)(&A[(size_t)(m0 + row + 64 * i) * K + k0 + kc]);
            }
            {
                v8bf val = {};
                if (wok)
                    val = *(const v8bf*)(&W[(size_t)(n0 + row) * K + k0 + kc]);
                *(v8bf*)(&Ws[0][row * LSTR + kc]) = val;
            }
            __syncthreads();
            Frag a0, a1, b0, b1;
            {
                int r = wm * 32 + fm;
                a0.h[0] = *(const v8bf*)(&As[0][r * LSTR + kb]);
                a0.h[1] = *(const v8bf*)(&As[0][r * LSTR + 16 + kb]);
                r += 16;
                a1.h[0] = *(const v8bf*)(&As[0][r * LSTR + kb]);
                a1.h[1] = *(const v8bf*)(&As[0][r * LSTR + 16 + kb]);
            }
            {
                int r = wn * 32 + fm;
                b0.h[0] = *(const v8bf*)(&Ws[0][r * LSTR + kb]);
                b0.h[1] = *(const v8bf*)(&Ws[0][r * LSTR + 16 + kb]);
                r += 16;
                b1.h[0] = *(const v8bf*)(&Ws[0][r * LSTR + kb]);
                b1.h[1] = *(const v8bf*)(&Ws[0][r * LSTR + 16 + kb]);
            }
            acc00 = __builtin_amdgcn_wmma_f32_16x16x32_bf16(false, a0.v, false, b0.v,
                                                            (short)0, acc00, false, false);
            acc01 = __builtin_amdgcn_wmma_f32_16x16x32_bf16(false, a0.v, false, b1.v,
                                                            (short)0, acc01, false, false);
            acc10 = __builtin_amdgcn_wmma_f32_16x16x32_bf16(false, a1.v, false, b0.v,
                                                            (short)0, acc10, false, false);
            acc11 = __builtin_amdgcn_wmma_f32_16x16x32_bf16(false, a1.v, false, b1.v,
                                                            (short)0, acc11, false, false);
            __syncthreads();
        }
    }

    // Store: D element (m,n) lives in VGPR (m&7), lane (n&15) + 16*((m>>3)&1)
    const int cn = lane & 15;
    const int rh = (lane >> 4) * 8;
#pragma unroll
    for (int ij = 0; ij < 4; ++ij) {
        const int i = ij >> 1, j = ij & 1;
        v8f accv = (ij == 0) ? acc00 : (ij == 1) ? acc01 : (ij == 2) ? acc10 : acc11;
        int n = n0 + wn * 32 + j * 16 + cn;
        if (NGUARD && n >= N) continue;
        float bv = bias ? bias[n] : 0.0f;
#pragma unroll
        for (int r = 0; r < 8; ++r) {
            int m = m0 + wm * 32 + i * 16 + rh + r;
            size_t idx = (size_t)m * N + n;
            float v = accv[r] + bv;
            if (accumulate) C[idx] += v; else C[idx] = v;
        }
    }
}

// ---------------------------------------------------------------------------
// LayerNorm over D_MODEL=256, one wave per row, bf16 output (feeds WMMA)
// ---------------------------------------------------------------------------
__global__ __launch_bounds__(256)
void layernorm_bf16(const float* __restrict__ h, bf16_t* __restrict__ out,
                    const float* __restrict__ scale, const float* __restrict__ bias)
{
    int row  = blockIdx.x * 8 + (threadIdx.x >> 5);
    int lane = threadIdx.x & 31;
    const float* p = h + (size_t)row * D_MODEL;
    float v[8];
    float s = 0.0f;
#pragma unroll
    for (int i = 0; i < 8; ++i) { v[i] = p[lane + 32 * i]; s += v[i]; }
#pragma unroll
    for (int o = 16; o > 0; o >>= 1) s += __shfl_xor(s, o, 32);
    float mu  = s * (1.0f / D_MODEL);
    float var = 0.0f;
#pragma unroll
    for (int i = 0; i < 8; ++i) { float d = v[i] - mu; var += d * d; }
#pragma unroll
    for (int o = 16; o > 0; o >>= 1) var += __shfl_xor(var, o, 32);
    float r = rsqrtf(var * (1.0f / D_MODEL) + 1e-5f);
#pragma unroll
    for (int i = 0; i < 8; ++i) {
        int c = lane + 32 * i;
        out[(size_t)row * D_MODEL + c] = (bf16_t)((v[i] - mu) * r * scale[c] + bias[c]);
    }
}

// ---------------------------------------------------------------------------
// Depthwise causal conv(4) + SiLU over xc half of xz.
// grid (T/256, B), 256 threads; thread owns 2 channels, register window.
// ---------------------------------------------------------------------------
__global__ __launch_bounds__(256)
void conv_silu(const float* __restrict__ xz, const float* __restrict__ cw,
               const float* __restrict__ cb, float* __restrict__ xcf,
               bf16_t* __restrict__ xcb)
{
    const int b  = blockIdx.y;
    const int t0 = blockIdx.x * 256;
#pragma unroll
    for (int half = 0; half < 2; ++half) {
        const int d = threadIdx.x + half * 256;
        const float w0 = cw[d * 4 + 0], w1 = cw[d * 4 + 1];
        const float w2 = cw[d * 4 + 2], w3 = cw[d * 4 + 3];
        const float bv = cb[d];
        const size_t base = (size_t)b * T_LEN * (2 * D_INNER) + d;
        float xm3 = (t0 - 3 >= 0) ? xz[base + (size_t)(t0 - 3) * (2 * D_INNER)] : 0.0f;
        float xm2 = (t0 - 2 >= 0) ? xz[base + (size_t)(t0 - 2) * (2 * D_INNER)] : 0.0f;
        float xm1 = (t0 - 1 >= 0) ? xz[base + (size_t)(t0 - 1) * (2 * D_INNER)] : 0.0f;
        for (int tt = 0; tt < 256; ++tt) {
            const int t = t0 + tt;
            float x0 = xz[base + (size_t)t * (2 * D_INNER)];
            float c  = bv + w0 * xm3 + w1 * xm2 + w2 * xm1 + w3 * x0;
            float sv = c / (1.0f + __expf(-c));
            size_t oi = ((size_t)b * T_LEN + t) * D_INNER + d;
            xcf[oi] = sv;
            xcb[oi] = (bf16_t)sv;
            xm3 = xm2; xm2 = xm1; xm1 = x0;
        }
    }
}

// ---------------------------------------------------------------------------
// Selective scan, fused: dt_proj + softplus + recurrence + C-reduce + D-skip +
// SiLU(z) gating. grid (D_INNER/128, B), 128 threads; thread owns one channel.
// x_dbl rows (48 f32, shared by all channels of a batch) staged via LDS.
// ---------------------------------------------------------------------------
__global__ __launch_bounds__(128)
void selective_scan(const float* __restrict__ x_dbl,  // (BT, 48)
                    const float* __restrict__ xc,     // (BT, 512) f32
                    const float* __restrict__ xz,     // (BT, 1024), z = cols [512,1024)
                    const float* __restrict__ dtw,    // (512, 16)
                    const float* __restrict__ dtb,    // (512)
                    const float* __restrict__ A_log,  // (512, 16)
                    const float* __restrict__ Dp,     // (512)
                    bf16_t* __restrict__ ybf)         // (BT, 512)
{
    const int b = blockIdx.y;
    const int d = blockIdx.x * 128 + threadIdx.x;

    float w[D_STATE], A[D_STATE], h[D_STATE];
#pragma unroll
    for (int n = 0; n < D_STATE; ++n) {
        w[n] = dtw[d * DT_RANK + n];
        A[n] = -__expf(A_log[d * D_STATE + n]);
        h[n] = 0.0f;
    }
    const float db = dtb[d], Dd = Dp[d];

    __shared__ float sx[64 * XD];   // 64 timesteps of x_dbl = 12 KB
    for (int tc = 0; tc < T_LEN; tc += 64) {
        __syncthreads();
        for (int i = threadIdx.x; i < 64 * XD; i += 128)
            sx[i] = x_dbl[((size_t)b * T_LEN + tc) * XD + i];
        __syncthreads();
        for (int tt = 0; tt < 64; ++tt) {
            const int t = tc + tt;
            const float* xr = &sx[tt * XD];     // broadcast reads, conflict-free
            float dtv = db;
#pragma unroll
            for (int n = 0; n < DT_RANK; ++n) dtv += xr[n] * w[n];
            dtv = (dtv > 20.0f) ? dtv : log1pf(__expf(dtv));   // softplus
            const float xt = xc[((size_t)b * T_LEN + t) * D_INNER + d];
            float y = 0.0f;
#pragma unroll
            for (int n = 0; n < D_STATE; ++n) {
                h[n] = h[n] * __expf(dtv * A[n]) + dtv * xr[DT_RANK + n] * xt;
                y += h[n] * xr[DT_RANK + D_STATE + n];
            }
            y += Dd * xt;
            const float z = xz[((size_t)b * T_LEN + t) * (2 * D_INNER) + D_INNER + d];
            y *= z / (1.0f + __expf(-z));       // * silu(z)
            ybf[((size_t)b * T_LEN + t) * D_INNER + d] = (bf16_t)y;
        }
    }
}

// ---------------------------------------------------------------------------
// Host orchestration
// ---------------------------------------------------------------------------
static inline size_t align256(size_t x) { return (x + 255) & ~(size_t)255; }

extern "C" void kernel_launch(void* const* d_in, const int* in_sizes, int n_in,
                              void* d_out, int out_size, void* d_ws, size_t ws_size,
                              hipStream_t stream)
{
    (void)in_sizes; (void)n_in; (void)out_size; (void)ws_size;
    const float* x          = (const float*)d_in[0];
    const float* emb_w      = (const float*)d_in[1];
    const float* emb_b      = (const float*)d_in[2];
    const float* norm_scale = (const float*)d_in[3];
    const float* norm_bias  = (const float*)d_in[4];
    const float* in_proj_w  = (const float*)d_in[5];
    const float* conv_w     = (const float*)d_in[6];
    const float* conv_b     = (const float*)d_in[7];
    const float* x_proj_w   = (const float*)d_in[8];
    const float* dt_proj_w  = (const float*)d_in[9];
    const float* dt_proj_b  = (const float*)d_in[10];
    const float* A_log      = (const float*)d_in[11];
    const float* Dparam     = (const float*)d_in[12];
    const float* out_proj_w = (const float*)d_in[13];
    const float* fc_w       = (const float*)d_in[14];
    const float* fc_b       = (const float*)d_in[15];

    char* ws = (char*)d_ws;
    size_t off = 0;
    float*  hidden = (float*)(ws + off);  off += align256((size_t)BT * D_MODEL * 4);
    bf16_t* normed = (bf16_t*)(ws + off); off += align256((size_t)BT * D_MODEL * 2);
    float*  xz     = (float*)(ws + off);  off += align256((size_t)BT * 2 * D_INNER * 4);
    float*  xcf    = (float*)(ws + off);  off += align256((size_t)BT * D_INNER * 4);
    bf16_t* xcb    = (bf16_t*)(ws + off); off += align256((size_t)BT * D_INNER * 2);
    float*  xdbl   = (float*)(ws + off);  off += align256((size_t)BT * XD * 4);
    bf16_t* ybf    = (bf16_t*)(ws + off); off += align256((size_t)BT * D_INNER * 2);
    bf16_t* xbf    = (bf16_t*)(ws + off); off += align256((size_t)BT * 64 * 2);
    bf16_t* wb     = (bf16_t*)(ws + off);

    bf16_t* wb_emb = wb;                               // 256*64
    bf16_t* wb_in  = wb_emb + 256 * 64;                // 4*1024*256
    bf16_t* wb_xp  = wb_in  + N_LAYERS * 1024 * 256;   // 4*48*512
    bf16_t* wb_op  = wb_xp  + N_LAYERS * XD * 512;     // 4*256*512
    bf16_t* wb_fc  = wb_op  + N_LAYERS * 256 * 512;    // 32*256

    auto cvt = [&](const float* s, bf16_t* d, int n) {
        cvt_f32_bf16<<<(n + 255) / 256, 256, 0, stream>>>(s, d, n);
    };
    cvt(x, xbf, BT * 64);
    cvt(emb_w, wb_emb, 256 * 64);
    cvt(in_proj_w, wb_in, N_LAYERS * 1024 * 256);
    cvt(x_proj_w, wb_xp, N_LAYERS * XD * 512);
    cvt(out_proj_w, wb_op, N_LAYERS * 256 * 512);
    cvt(fc_w, wb_fc, N_CLASSES * 256);

    const dim3 blk(256);
    const int MB = BT / 128;   // 256 M-tiles

    // Embedding: hidden = x @ emb_w^T + emb_b   (M=32768, N=256, K=64)
    gemm_bf16_wmma<false><<<dim3(256 / 64, MB), blk, 0, stream>>>(
        xbf, wb_emb, hidden, emb_b, BT, D_MODEL, 64, 0);

    for (int l = 0; l < N_LAYERS; ++l) {
        layernorm_bf16<<<BT / 8, 256, 0, stream>>>(
            hidden, normed, norm_scale + l * D_MODEL, norm_bias + l * D_MODEL);

        // xz = normed @ in_proj_w^T   (N=1024, K=256)
        gemm_bf16_wmma<false><<<dim3(1024 / 64, MB), blk, 0, stream>>>(
            normed, wb_in + (size_t)l * 1024 * 256, xz, nullptr, BT, 1024, 256, 0);

        conv_silu<<<dim3(T_LEN / 256, B_SZ), blk, 0, stream>>>(
            xz, conv_w + (size_t)l * D_INNER * 4, conv_b + (size_t)l * D_INNER, xcf, xcb);

        // x_dbl = xc @ x_proj_w^T    (N=48, K=512)
        gemm_bf16_wmma<true><<<dim3(1, MB), blk, 0, stream>>>(
            xcb, wb_xp + (size_t)l * XD * 512, xdbl, nullptr, BT, XD, 512, 0);

        selective_scan<<<dim3(D_INNER / 128, B_SZ), dim3(128), 0, stream>>>(
            xdbl, xcf, xz,
            dt_proj_w + (size_t)l * D_INNER * DT_RANK,
            dt_proj_b + (size_t)l * D_INNER,
            A_log + (size_t)l * D_INNER * D_STATE,
            Dparam + (size_t)l * D_INNER, ybf);

        // hidden += y @ out_proj_w^T  (N=256, K=512, residual accumulate)
        gemm_bf16_wmma<false><<<dim3(256 / 64, MB), blk, 0, stream>>>(
            ybf, wb_op + (size_t)l * 256 * 512, hidden, nullptr, BT, D_MODEL, 512, 1);
    }

    // logits = hidden @ fc_w^T + fc_b   (N=32, K=256)
    cvt(hidden, normed, BT * D_MODEL);   // reuse normed buffer as bf16 hidden
    gemm_bf16_wmma<true><<<dim3(1, MB), blk, 0, stream>>>(
        normed, wb_fc, (float*)d_out, fc_b, BT, N_CLASSES, 256, 0);
}
